// Head_26723286516320
// MI455X (gfx1250) — matrix operand order repaired and checked by reading, not compiled, not run on hardware
//
#include <hip/hip_runtime.h>

#define B_DIM 128
#define T_DIM 256
#define D_DIM 1024
#define LDSS  40   // padded LDS row stride (bf16 elems): 80B rows -> conflict-free b128 frag loads

#ifndef __has_builtin
#define __has_builtin(x) 0
#endif
#if __has_builtin(__builtin_amdgcn_global_load_async_to_lds_b128)
#define HAS_ASYNC_LDS 1
#else
#define HAS_ASYNC_LDS 0
#endif
#if __has_builtin(__builtin_amdgcn_s_wait_asynccnt)
#define HAS_WAIT_ASYNC 1
#else
#define HAS_WAIT_ASYNC 0
#endif

typedef __attribute__((ext_vector_type(16))) __bf16 v16bf;
typedef __attribute__((ext_vector_type(8)))  float  v8f;
typedef int v4i_vs __attribute__((__vector_size__(16)));   // matches builtin param type

union BFrag { v16bf v; uint4 q[2]; };

// 16-byte global -> LDS tile copy. Async (ASYNCcnt, no VGPR round-trip) when the
// toolchain exposes the gfx1250 builtin; plain copy otherwise.
__device__ __forceinline__ void cp16(__bf16* l, const __bf16* g) {
#if HAS_ASYNC_LDS
  __builtin_amdgcn_global_load_async_to_lds_b128(
      (__attribute__((address_space(1))) v4i_vs*)(unsigned long long)(const void*)g,
      (__attribute__((address_space(3))) v4i_vs*)l, 0, 0);
#else
  *(uint4*)l = *(const uint4*)g;
#endif
}

template <int N>
__device__ __forceinline__ void wait_async() {
#if HAS_ASYNC_LDS
#if HAS_WAIT_ASYNC
  __builtin_amdgcn_s_wait_asynccnt(N);
#else
  asm volatile("s_wait_asynccnt %0" ::"i"(N) : "memory");
#endif
#endif
}

__device__ __forceinline__ v8f wmma_bf16(v16bf a, v16bf b, v8f c) {
  // (neg_a, A, neg_b, B, c_mod, C, reuse_a, reuse_b)
  return __builtin_amdgcn_wmma_f32_16x16x32_bf16(false, a, false, b, (short)0, c,
                                                 false, false);
}

__device__ __forceinline__ v8f vzero8() {
  v8f z = {0.f, 0.f, 0.f, 0.f, 0.f, 0.f, 0.f, 0.f};
  return z;
}

// A fragment 16x32 bf16 from row-major LDS tile (stride LDSS).
// lane l: row = l%16; low half holds K 0..7 & 16..23, high half K 8..15 & 24..31.
__device__ __forceinline__ v16bf frag_a(const __bf16* base, int lane) {
  int row = lane & 15;
  int k0  = (lane >> 4) << 3;
  BFrag f;
  f.q[0] = *(const uint4*)(base + row * LDSS + k0);
  f.q[1] = *(const uint4*)(base + row * LDSS + k0 + 16);
  return f.v;
}

// B fragment 32x16 bf16; source tile stored [n][k] (column n of B contiguous in k).
// lane l: col = l%16; low half holds K 0..7 & 16..23, high half K 8..15 & 24..31.
__device__ __forceinline__ v16bf frag_b(const __bf16* base, int lane) {
  int col = lane & 15;
  int k0  = (lane >> 4) << 3;
  BFrag f;
  f.q[0] = *(const uint4*)(base + col * LDSS + k0);
  f.q[1] = *(const uint4*)(base + col * LDSS + k0 + 16);
  return f.v;
}

// ---------------------------------------------------------------------------
// Kernel 0: streaming f32 -> bf16 conversion (x and the three weight matrices),
// so the GEMM kernels can stage tiles with pure async b128 copies.
// ---------------------------------------------------------------------------
__global__ __launch_bounds__(256) void cvt_f32_bf16(const float* __restrict__ src,
                                                    __bf16* __restrict__ dst, int n4) {
  int i = blockIdx.x * blockDim.x + threadIdx.x;
  if (i < n4) {
    float4 f = ((const float4*)src)[i];
    union { __bf16 h[4]; uint2 u; } o;
    o.h[0] = (__bf16)f.x; o.h[1] = (__bf16)f.y;
    o.h[2] = (__bf16)f.z; o.h[3] = (__bf16)f.w;
    ((uint2*)dst)[i] = o.u;
  }
}

// ---------------------------------------------------------------------------
// Kernel 1: fused QKV projection GEMM.  y = x @ W.T + b (bf16 in, f32 acc).
// M = B*T, N = K = D.  blockIdx.z selects Q/K/V; V stored transposed [B,D,T].
// Double-buffered async tile staging overlapped with WMMA.
// ---------------------------------------------------------------------------
__global__ __launch_bounds__(256) void qkv_gemm(
    const __bf16* __restrict__ xb, const __bf16* __restrict__ Wb,
    const float* __restrict__ bq, const float* __restrict__ bk,
    const float* __restrict__ bv,
    __bf16* __restrict__ Qg, __bf16* __restrict__ Kg, __bf16* __restrict__ VTg)
{
  __shared__ __align__(16) __bf16 As[2][128 * LDSS];
  __shared__ __align__(16) __bf16 Bs[2][128 * LDSS];
  const int tid   = threadIdx.x;
  const int lane  = tid & 31;
  const int wave  = tid >> 5;
  const int waveM = wave >> 1;           // 0..3 (32 rows each)
  const int waveN = wave & 1;            // 0..1 (64 cols each)
  const int n0 = blockIdx.x * 128;       // N fastest -> 2MB weight stripe stays in L2
  const int m0 = blockIdx.y * 128;
  const int z  = blockIdx.z;             // 0=Q 1=K 2=V
  const __bf16* W   = Wb + (size_t)z * D_DIM * D_DIM;
  const float* bias = (z == 0) ? bq : (z == 1) ? bk : bv;

  const int cr = tid >> 2;               // 0..63
  const int cc = (tid & 3) * 8;          // 0/8/16/24
  auto stage = [&](int it, int buf) {    // 4 async b128 per thread
    int kk = it * 32;
#pragma unroll
    for (int i = 0; i < 2; ++i) {
      int r = cr + i * 64;
      cp16(&As[buf][r * LDSS + cc], xb + (size_t)(m0 + r) * D_DIM + kk + cc);
      cp16(&Bs[buf][r * LDSS + cc], W  + (size_t)(n0 + r) * D_DIM + kk + cc);
    }
  };

  v8f acc[2][4];
#pragma unroll
  for (int i = 0; i < 2; ++i)
#pragma unroll
    for (int j = 0; j < 4; ++j) acc[i][j] = vzero8();

  const int NT = D_DIM / 32;
  stage(0, 0);
  for (int it = 0; it < NT; ++it) {
    const int cur = it & 1;
    if (it + 1 < NT) { stage(it + 1, cur ^ 1); wait_async<4>(); }
    else             { wait_async<0>(); }
    __syncthreads();                     // tile `it` visible to all waves
#pragma unroll
    for (int mh = 0; mh < 2; ++mh) {
      v16bf a = frag_a(&As[cur][(waveM * 32 + mh * 16) * LDSS], lane);
#pragma unroll
      for (int nf = 0; nf < 4; ++nf) {
        v16bf b = frag_b(&Bs[cur][(waveN * 64 + nf * 16) * LDSS], lane);
        acc[mh][nf] = wmma_bf16(a, b, acc[mh][nf]);
      }
    }
    __syncthreads();                     // done reading before tile it+2 overwrites
  }

  const int lcol = lane & 15, lhi = lane >> 4;
  if (z < 2) {
    __bf16* O = (z == 0) ? Qg : Kg;
#pragma unroll
    for (int mh = 0; mh < 2; ++mh) {
#pragma unroll
      for (int nf = 0; nf < 4; ++nf) {
        int col  = n0 + waveN * 64 + nf * 16 + lcol;
        float bb = bias[col];
        int rowb = m0 + waveM * 32 + mh * 16 + lhi * 8;
#pragma unroll
        for (int v = 0; v < 8; ++v)
          O[(size_t)(rowb + v) * D_DIM + col] = (__bf16)(acc[mh][nf][v] + bb);
      }
    }
  } else {                               // V: store transposed [B, D, T]
#pragma unroll
    for (int mh = 0; mh < 2; ++mh) {
#pragma unroll
      for (int nf = 0; nf < 4; ++nf) {
        int col  = n0 + waveN * 64 + nf * 16 + lcol;
        float bb = bias[col];
        int rowb = m0 + waveM * 32 + mh * 16 + lhi * 8;
#pragma unroll
        for (int v = 0; v < 8; ++v) {
          int row  = rowb + v;
          int bidx = row >> 8;           // / T
          int t    = row & (T_DIM - 1);
          VTg[((size_t)bidx * D_DIM + col) * T_DIM + t] = (__bf16)(acc[mh][nf][v] + bb);
        }
      }
    }
  }
}

// ---------------------------------------------------------------------------
// Kernel 2: scores = scale * Q K^T, causal mask, softmax -> bf16 probs [B,T,T].
// Each wave owns 16 query rows x all 256 key cols; causal skip halves WMMA work.
// ---------------------------------------------------------------------------
__global__ __launch_bounds__(256) void attn_scores_softmax(
    const __bf16* __restrict__ Qg, const __bf16* __restrict__ Kg,
    __bf16* __restrict__ P)
{
  __shared__ __align__(16) __bf16 Qs[2][128 * LDSS];
  __shared__ __align__(16) __bf16 Ks[2][256 * LDSS];
  const int tid = threadIdx.x, lane = tid & 31, wave = tid >> 5;
  const int b  = blockIdx.y;
  const int q0 = blockIdx.x * 128;
  const int qw = q0 + wave * 16;          // wave's query-row base (within T)
  const int nfrags = (qw >> 4) + 1;       // causal: s-frag base must be <= qw+15

  const int cr = tid >> 2;
  const int cc = (tid & 3) * 8;
  auto stage = [&](int it, int buf) {     // 6 async b128 per thread
    int dd = it * 32;
#pragma unroll
    for (int i = 0; i < 2; ++i) {
      int r = cr + i * 64;
      cp16(&Qs[buf][r * LDSS + cc],
           Qg + (size_t)(b * T_DIM + q0 + r) * D_DIM + dd + cc);
    }
#pragma unroll
    for (int i = 0; i < 4; ++i) {
      int r = cr + i * 64;
      cp16(&Ks[buf][r * LDSS + cc],
           Kg + (size_t)(b * T_DIM + r) * D_DIM + dd + cc);
    }
  };

  v8f acc[16];
#pragma unroll
  for (int i = 0; i < 16; ++i) acc[i] = vzero8();

  const int NT = D_DIM / 32;
  stage(0, 0);
  for (int it = 0; it < NT; ++it) {
    const int cur = it & 1;
    if (it + 1 < NT) { stage(it + 1, cur ^ 1); wait_async<6>(); }
    else             { wait_async<0>(); }
    __syncthreads();
    v16bf a = frag_a(&Qs[cur][(wave * 16) * LDSS], lane);
#pragma unroll
    for (int nf = 0; nf < 16; ++nf) {
      if (nf < nfrags) {                  // wave-uniform branch
        v16bf kb = frag_b(&Ks[cur][(nf * 16) * LDSS], lane);
        acc[nf] = wmma_bf16(a, kb, acc[nf]);
      }
    }
    __syncthreads();
  }

  const float scale = 0.03125f;           // 1/sqrt(1024)
  const int lcol = lane & 15, lhi = lane >> 4;
  float rmax[8], rsum[8];
#pragma unroll
  for (int v = 0; v < 8; ++v) rmax[v] = -3.0e38f;

#pragma unroll
  for (int nf = 0; nf < 16; ++nf) {
    if (nf < nfrags) {
#pragma unroll
      for (int v = 0; v < 8; ++v) {
        int t = qw + lhi * 8 + v;
        int s = nf * 16 + lcol;
        float val = acc[nf][v] * scale;
        if (s > t) val = -3.0e38f;        // causal mask on the diagonal frag
        acc[nf][v] = val;
        rmax[v] = fmaxf(rmax[v], val);
      }
    }
  }
  // Row reductions across 16-lane half-groups (rows 0-7 in lanes 0-15,
  // rows 8-15 in lanes 16-31); xor masks 1,2,4,8 stay inside each group.
#pragma unroll
  for (int v = 0; v < 8; ++v) {
    float m = rmax[v];
    m = fmaxf(m, __shfl_xor(m, 1, 32));
    m = fmaxf(m, __shfl_xor(m, 2, 32));
    m = fmaxf(m, __shfl_xor(m, 4, 32));
    m = fmaxf(m, __shfl_xor(m, 8, 32));
    rmax[v] = m;
    rsum[v] = 0.f;
  }
#pragma unroll
  for (int nf = 0; nf < 16; ++nf) {
    if (nf < nfrags) {
#pragma unroll
      for (int v = 0; v < 8; ++v) {
        float e = __expf(acc[nf][v] - rmax[v]);
        acc[nf][v] = e;
        rsum[v] += e;
      }
    }
  }
#pragma unroll
  for (int v = 0; v < 8; ++v) {
    float s = rsum[v];
    s += __shfl_xor(s, 1, 32);
    s += __shfl_xor(s, 2, 32);
    s += __shfl_xor(s, 4, 32);
    s += __shfl_xor(s, 8, 32);
    rsum[v] = 1.0f / s;
  }
  const size_t rowbase = ((size_t)b * T_DIM + qw) * T_DIM;
#pragma unroll
  for (int nf = 0; nf < 16; ++nf) {
#pragma unroll
    for (int v = 0; v < 8; ++v) {
      float p = (nf < nfrags) ? acc[nf][v] * rsum[v] : 0.0f;  // zero-fill s>t
      P[rowbase + (size_t)(lhi * 8 + v) * T_DIM + nf * 16 + lcol] = (__bf16)p;
    }
  }
}

// ---------------------------------------------------------------------------
// Kernel 3: out = P @ V per batch.  A = probs bf16 [T,T], B = V^T bf16 [D,T].
// f32 output [B,T,D].
// ---------------------------------------------------------------------------
__global__ __launch_bounds__(256) void pv_gemm(
    const __bf16* __restrict__ P, const __bf16* __restrict__ VTg,
    float* __restrict__ out)
{
  __shared__ __align__(16) __bf16 As[2][128 * LDSS];
  __shared__ __align__(16) __bf16 Bs[2][128 * LDSS];
  const int tid = threadIdx.x, lane = tid & 31, wave = tid >> 5;
  const int waveM = wave >> 1, waveN = wave & 1;
  const int n0 = blockIdx.x * 128;        // dout tile
  const int m0 = blockIdx.y * 128;        // t tile
  const int b  = blockIdx.z;

  const int cr = tid >> 2;
  const int cc = (tid & 3) * 8;
  auto stage = [&](int it, int buf) {     // 4 async b128 per thread
    int kk = it * 32;
#pragma unroll
    for (int i = 0; i < 2; ++i) {
      int r = cr + i * 64;
      cp16(&As[buf][r * LDSS + cc],
           P + ((size_t)b * T_DIM + m0 + r) * T_DIM + kk + cc);
      cp16(&Bs[buf][r * LDSS + cc],
           VTg + ((size_t)b * D_DIM + n0 + r) * T_DIM + kk + cc);
    }
  };

  v8f acc[2][4];
#pragma unroll
  for (int i = 0; i < 2; ++i)
#pragma unroll
    for (int j = 0; j < 4; ++j) acc[i][j] = vzero8();

  const int NT = T_DIM / 32;
  stage(0, 0);
  for (int it = 0; it < NT; ++it) {
    const int cur = it & 1;
    if (it + 1 < NT) { stage(it + 1, cur ^ 1); wait_async<4>(); }
    else             { wait_async<0>(); }
    __syncthreads();
#pragma unroll
    for (int mh = 0; mh < 2; ++mh) {
      v16bf a = frag_a(&As[cur][(waveM * 32 + mh * 16) * LDSS], lane);
#pragma unroll
      for (int nf = 0; nf < 4; ++nf) {
        v16bf bb = frag_b(&Bs[cur][(waveN * 64 + nf * 16) * LDSS], lane);
        acc[mh][nf] = wmma_bf16(a, bb, acc[mh][nf]);
      }
    }
    __syncthreads();
  }

  const int lcol = lane & 15, lhi = lane >> 4;
#pragma unroll
  for (int mh = 0; mh < 2; ++mh) {
#pragma unroll
    for (int nf = 0; nf < 4; ++nf) {
      int col  = n0 + waveN * 64 + nf * 16 + lcol;
      int rowb = m0 + waveM * 32 + mh * 16 + lhi * 8;
#pragma unroll
      for (int v = 0; v < 8; ++v)
        out[((size_t)b * T_DIM + rowb + v) * D_DIM + col] = acc[mh][nf][v];
    }
  }
}

// ---------------------------------------------------------------------------
extern "C" void kernel_launch(void* const* d_in, const int* in_sizes, int n_in,
                              void* d_out, int out_size, void* d_ws, size_t ws_size,
                              hipStream_t stream) {
  (void)in_sizes; (void)n_in; (void)out_size; (void)ws_size;
  const float* x  = (const float*)d_in[0];
  const float* Wk = (const float*)d_in[1];
  const float* bk = (const float*)d_in[2];
  const float* Wq = (const float*)d_in[3];
  const float* bq = (const float*)d_in[4];
  const float* Wv = (const float*)d_in[5];
  const float* bv = (const float*)d_in[6];
  float* out = (float*)d_out;

  const size_t xt = (size_t)B_DIM * T_DIM * D_DIM;   // 33.5M elems
  const size_t dd = (size_t)D_DIM * D_DIM;           // 1M elems
  __bf16* xb  = (__bf16*)d_ws;
  __bf16* Wb  = xb + xt;            // [3][D*D]: z order Q,K,V
  __bf16* Qg  = Wb + 3 * dd;
  __bf16* Kg  = Qg + xt;
  __bf16* VTg = Kg + xt;
  __bf16* P   = VTg + xt;           // [B,T,T] bf16 probs (16.8 MB, L2-resident)

  cvt_f32_bf16<<<(int)(xt / 4 + 255) / 256, 256, 0, stream>>>(x, xb, (int)(xt / 4));
  cvt_f32_bf16<<<(int)(dd / 4 + 255) / 256, 256, 0, stream>>>(Wq, Wb, (int)(dd / 4));
  cvt_f32_bf16<<<(int)(dd / 4 + 255) / 256, 256, 0, stream>>>(Wk, Wb + dd, (int)(dd / 4));
  cvt_f32_bf16<<<(int)(dd / 4 + 255) / 256, 256, 0, stream>>>(Wv, Wb + 2 * dd, (int)(dd / 4));

  qkv_gemm<<<dim3(D_DIM / 128, (B_DIM * T_DIM) / 128, 3), 256, 0, stream>>>(
      xb, Wb, bq, bk, bv, Qg, Kg, VTg);
  attn_scores_softmax<<<dim3(T_DIM / 128, B_DIM), 256, 0, stream>>>(Qg, Kg, P);
  pv_gemm<<<dim3(D_DIM / 128, T_DIM / 128, B_DIM), 256, 0, stream>>>(P, VTg, out);
}
